// INR_MoE_78262894068194
// MI455X (gfx1250) — matrix operand order repaired and checked by reading, not compiled, not run on hardware
//
#include <hip/hip_runtime.h>
#include <hip/hip_bf16.h>

typedef __attribute__((ext_vector_type(16))) _Float16 v16h;
typedef __attribute__((ext_vector_type(8)))  _Float16 v8h;
typedef __attribute__((ext_vector_type(8)))  float    v8f;

// Sizes fixed by the reference
#define E_EXP   8
#define D_IN    3
#define NLAYER  3
#define NTHREADS 64            // 2 waves; each wave owns 2 x 16-row A tiles
#define ROWS_WG  64

// ---------------------------------------------------------------------------
// Repack f32 weights [K=256][N=256] into f16 WMMA B-fragment order, with the
// K-slice outermost so each K=32 slice is a contiguous 16 KB block:
//   frag[kc][nc][lane][e]  (halfs), lane: col = lane&15, k = kc*32 + (lane&16) + e
// 24 hidden matrices (L*E) followed by M1 => 25 * 65536 halfs.
// ---------------------------------------------------------------------------
__global__ void repack_kernel(const float* __restrict__ Whid,
                              const float* __restrict__ M1,
                              _Float16* __restrict__ frag)
{
    const unsigned total = 25u * 65536u;
    for (unsigned o = blockIdx.x * blockDim.x + threadIdx.x; o < total;
         o += gridDim.x * blockDim.x) {
        unsigned m    = o >> 16;
        unsigned r    = o & 65535u;
        unsigned e    = r & 15u;
        unsigned lane = (r >> 4) & 31u;
        unsigned nc   = (r >> 9) & 15u;
        unsigned kc   = (r >> 13) & 7u;
        unsigned k    = kc * 32u + (lane & 16u) + e;
        unsigned col  = nc * 16u + (lane & 15u);
        const float* src = (m < 24u) ? (Whid + (size_t)m * 65536u) : M1;
        frag[o] = (_Float16)src[k * 256u + col];
    }
}

// ---------------------------------------------------------------------------
// Async-DMA one contiguous 16 KB K-slice of weight fragments into LDS.
// 64 threads x 256 linear bytes (16 x b128 each). INST_OFFSET applies to both
// the LDS and global addresses, so one base VGPR each covers all transfers.
// Tracked on ASYNCcnt.
// ---------------------------------------------------------------------------
__device__ __forceinline__ void stage_async(const _Float16* __restrict__ gsrc,
                                            _Float16* lbuf, int tid)
{
    // low 32 bits of a generic LDS pointer are the LDS byte address
    unsigned lds = (unsigned)(uintptr_t)lbuf + (unsigned)(tid * 256);
    const char* g = (const char*)gsrc + tid * 256;
    asm volatile(
        "global_load_async_to_lds_b128 %0, %1, off\n\t"
        "global_load_async_to_lds_b128 %0, %1, off offset:16\n\t"
        "global_load_async_to_lds_b128 %0, %1, off offset:32\n\t"
        "global_load_async_to_lds_b128 %0, %1, off offset:48\n\t"
        "global_load_async_to_lds_b128 %0, %1, off offset:64\n\t"
        "global_load_async_to_lds_b128 %0, %1, off offset:80\n\t"
        "global_load_async_to_lds_b128 %0, %1, off offset:96\n\t"
        "global_load_async_to_lds_b128 %0, %1, off offset:112\n\t"
        "global_load_async_to_lds_b128 %0, %1, off offset:128\n\t"
        "global_load_async_to_lds_b128 %0, %1, off offset:144\n\t"
        "global_load_async_to_lds_b128 %0, %1, off offset:160\n\t"
        "global_load_async_to_lds_b128 %0, %1, off offset:176\n\t"
        "global_load_async_to_lds_b128 %0, %1, off offset:192\n\t"
        "global_load_async_to_lds_b128 %0, %1, off offset:208\n\t"
        "global_load_async_to_lds_b128 %0, %1, off offset:224\n\t"
        "global_load_async_to_lds_b128 %0, %1, off offset:240\n\t"
        :: "v"(lds), "v"(g) : "memory");
}

__device__ __forceinline__ void wait_async_all()
{
    asm volatile("s_wait_asynccnt 0x0" ::: "memory");
}

// ---------------------------------------------------------------------------
// One 256->256 hidden layer on 64 rows using WMMA. 2 waves; each wave holds
// TWO 16-row A tiles so every B fragment read from LDS feeds two WMMAs
// (halves LDS B-traffic per WMMA, keeping it under the 256 B/clk WGP peak).
// Weights double-buffered via async DMA.
//   hbuf: activations f16, [wave][tile2][row16][256]  (32 KB)
//   wbuf0/1: one K=32 fragment slice each (16 KB each)
// ---------------------------------------------------------------------------
__device__ __forceinline__ void hidden_layer(_Float16* hbuf,
                                             _Float16* wbuf0, _Float16* wbuf1,
                                             const _Float16* __restrict__ wfrag,
                                             const float* __restrict__ bias,
                                             int tid)
{
    const int wid  = tid >> 5;
    const int lane = tid & 31;

    v8f zero = {0.f, 0.f, 0.f, 0.f, 0.f, 0.f, 0.f, 0.f};
    v8f acc[2][16];
#pragma unroll
    for (int t = 0; t < 2; ++t)
#pragma unroll
        for (int i = 0; i < 16; ++i) acc[t][i] = zero;

    // prologue: slice 0 -> wbuf0 (old readers of wbuf0 drained behind the
    // last barrier every wave has already passed)
    stage_async(wfrag, wbuf0, tid);

    for (int kc = 0; kc < 8; ++kc) {
        _Float16* cur = (kc & 1) ? wbuf1 : wbuf0;
        _Float16* nxt = (kc & 1) ? wbuf0 : wbuf1;
        wait_async_all();        // own DMA for slice kc complete
        __syncthreads();         // everyone's DMA done; nxt's old readers done
        if (kc < 7) stage_async(wfrag + (kc + 1) * 8192, nxt, tid);

        // Two A-fragments (two 16-row tiles) for this wave, K-slice kc:
        // lane<16: K = kc*32 + {0..15}; lane>=16: +16
        v16h a[2];
#pragma unroll
        for (int t = 0; t < 2; ++t) {
            const _Float16* hp = hbuf + wid * 8192 + t * 4096 +
                                 (lane & 15) * 256 + kc * 32 + (lane & 16);
            v8h lo = *(const v8h*)hp;
            v8h hi = *(const v8h*)(hp + 8);
            a[t] = __builtin_shufflevector(lo, hi, 0, 1, 2, 3, 4, 5, 6, 7,
                                           8, 9, 10, 11, 12, 13, 14, 15);
        }
#pragma unroll
        for (int nc = 0; nc < 16; ++nc) {
            v16h b = *(const v16h*)(cur + (nc * 32 + lane) * 16);
            acc[0][nc] = __builtin_amdgcn_wmma_f32_16x16x32_f16(
                false, a[0], false, b, (short)0, acc[0][nc], false, false);
            acc[1][nc] = __builtin_amdgcn_wmma_f32_16x16x32_f16(
                false, a[1], false, b, (short)0, acc[1][nc], false, false);
        }
    }

    // Epilogue: h = relu(acc + bias) back into this wave's tiles (own region,
    // LDS ops are in-order within a wave -> no barrier needed here).
    const int rbase = (lane & 16) >> 1;  // rows v or v+8
#pragma unroll
    for (int t = 0; t < 2; ++t) {
        _Float16* hw = hbuf + wid * 8192 + t * 4096;
#pragma unroll
        for (int nc = 0; nc < 16; ++nc) {
            int col = nc * 16 + (lane & 15);
            float bv = bias[col];
#pragma unroll
            for (int v = 0; v < 8; ++v) {
                float x = acc[t][nc][v] + bv;
                x = fmaxf(x, 0.f);
                hw[(rbase + v) * 256 + col] = (_Float16)x;
            }
        }
    }
}

// row -> hbuf offset for 64 rows laid out [wave][tile][row16][256]
__device__ __forceinline__ int hrow_off(int r)
{
    return (r >> 5) * 8192 + ((r >> 4) & 1) * 4096 + (r & 15) * 256;
}

// ---------------------------------------------------------------------------
// Decoder: all 8 experts densely. grid = (N/64, E), block = 64
// ---------------------------------------------------------------------------
__global__ __launch_bounds__(NTHREADS)
void decoder_kernel(const float* __restrict__ P,
                    const float* __restrict__ Win, const float* __restrict__ bin,
                    const float* __restrict__ bhid,
                    const _Float16* __restrict__ whidFrag,
                    const float* __restrict__ Wout, const float* __restrict__ bout,
                    float* __restrict__ pred, int N)
{
    __shared__ alignas(32) _Float16 hbuf[2 * 32 * 256];       // 32 KB
    __shared__ alignas(32) _Float16 wbuf[2][16 * 32 * 16];    // 2 x 16 KB

    const int tid  = threadIdx.x;
    const int e    = blockIdx.y;
    const int row0 = blockIdx.x * ROWS_WG;

    // ---- layer 1: [N,3] x [3,256], VALU; one row per thread ----
    {
        int gr = row0 + tid;
        float p0 = P[gr * 3 + 0], p1 = P[gr * 3 + 1], p2 = P[gr * 3 + 2];
        const float* w0 = Win + (e * 3 + 0) * 256;
        const float* w1 = Win + (e * 3 + 1) * 256;
        const float* w2 = Win + (e * 3 + 2) * 256;
        const float* bb = bin + e * 256;
        _Float16* hw = hbuf + hrow_off(tid);
#pragma unroll 4
        for (int j = 0; j < 256; ++j) {
            float x = fmaf(p0, w0[j], fmaf(p1, w1[j], fmaf(p2, w2[j], bb[j])));
            hw[j] = (_Float16)fmaxf(x, 0.f);
        }
    }

    // ---- 3 hidden 256x256 layers: WMMA + async double-buffered weights ----
    for (int l = 0; l < NLAYER; ++l) {
        hidden_layer(hbuf, wbuf[0], wbuf[1],
                     whidFrag + (size_t)(l * E_EXP + e) * 65536u,
                     bhid + (l * E_EXP + e) * 256, tid);
    }
    __syncthreads();

    // ---- output layer: [256] -> 1, VALU dot; one row per thread ----
    {
        int gr = row0 + tid;
        const _Float16* hp = hbuf + hrow_off(tid);
        const float* w = Wout + e * 256;
        float acc = bout[e];
#pragma unroll 8
        for (int k = 0; k < 256; ++k) acc = fmaf((float)hp[k], w[k], acc);
        pred[(size_t)e * N + gr] = acc;
    }
}

// ---------------------------------------------------------------------------
// Manager: 3->256->256->8, softmax(T=1), clamp 1e-5, argmax. grid = N/64
// ---------------------------------------------------------------------------
__global__ __launch_bounds__(NTHREADS)
void manager_kernel(const float* __restrict__ P,
                    const float* __restrict__ M0, const float* __restrict__ c0,
                    const _Float16* __restrict__ m1frag, const float* __restrict__ c1,
                    const float* __restrict__ M2, const float* __restrict__ c2,
                    float* __restrict__ qout, float* __restrict__ idxf,
                    int* __restrict__ idxi, int N)
{
    __shared__ alignas(32) _Float16 hbuf[2 * 32 * 256];
    __shared__ alignas(32) _Float16 wbuf[2][16 * 32 * 16];

    const int tid  = threadIdx.x;
    const int row0 = blockIdx.x * ROWS_WG;

    // layer 1 (M0: [3][256]); one row per thread
    {
        int gr = row0 + tid;
        float p0 = P[gr * 3 + 0], p1 = P[gr * 3 + 1], p2 = P[gr * 3 + 2];
        const float* w0 = M0;
        const float* w1 = M0 + 256;
        const float* w2 = M0 + 512;
        _Float16* hw = hbuf + hrow_off(tid);
#pragma unroll 4
        for (int j = 0; j < 256; ++j) {
            float x = fmaf(p0, w0[j], fmaf(p1, w1[j], fmaf(p2, w2[j], c0[j])));
            hw[j] = (_Float16)fmaxf(x, 0.f);
        }
    }

    hidden_layer(hbuf, wbuf[0], wbuf[1], m1frag, c1, tid);
    __syncthreads();

    {
        int gr = row0 + tid;
        const _Float16* hp = hbuf + hrow_off(tid);
        float q[E_EXP];
#pragma unroll
        for (int ex = 0; ex < E_EXP; ++ex) q[ex] = c2[ex];
        for (int k = 0; k < 256; ++k) {
            float h = (float)hp[k];
#pragma unroll
            for (int ex = 0; ex < E_EXP; ++ex)
                q[ex] = fmaf(h, M2[k * E_EXP + ex], q[ex]);
        }
        // softmax (T=1) -> clamp -> argmax (first max, strict >)
        float m = q[0];
#pragma unroll
        for (int ex = 1; ex < E_EXP; ++ex) m = fmaxf(m, q[ex]);
        float s = 0.f;
#pragma unroll
        for (int ex = 0; ex < E_EXP; ++ex) { q[ex] = __expf(q[ex] - m); s += q[ex]; }
        float inv = 1.f / s;
        int best = 0; float bv = -1.f;
#pragma unroll
        for (int ex = 0; ex < E_EXP; ++ex) {
            float v = fmaxf(q[ex] * inv, 1e-5f);
            qout[(size_t)gr * E_EXP + ex] = v;
            if (v > bv) { bv = v; best = ex; }
        }
        idxf[gr] = (float)best;
        idxi[gr] = best;
    }
}

// ---------------------------------------------------------------------------
// Gather the chosen expert's prediction: sel[n] = pred[idx[n], n, 0]
// ---------------------------------------------------------------------------
__global__ void select_kernel(const float* __restrict__ pred,
                              const int* __restrict__ idxi,
                              float* __restrict__ sel, int N)
{
    int n = blockIdx.x * blockDim.x + threadIdx.x;
    if (n < N) sel[n] = pred[(size_t)idxi[n] * N + n];
}

// ---------------------------------------------------------------------------
extern "C" void kernel_launch(void* const* d_in, const int* in_sizes, int n_in,
                              void* d_out, int out_size, void* d_ws, size_t ws_size,
                              hipStream_t stream)
{
    const float* P    = (const float*)d_in[0];
    const float* Win  = (const float*)d_in[1];
    const float* bin  = (const float*)d_in[2];
    const float* Whid = (const float*)d_in[3];
    const float* bhid = (const float*)d_in[4];
    const float* Wout = (const float*)d_in[5];
    const float* bout = (const float*)d_in[6];
    const float* M0   = (const float*)d_in[7];
    const float* c0   = (const float*)d_in[8];
    const float* M1   = (const float*)d_in[9];
    const float* c1   = (const float*)d_in[10];
    const float* M2   = (const float*)d_in[11];
    const float* c2   = (const float*)d_in[12];

    const int N = in_sizes[0] / D_IN;   // 32768

    // workspace: 25 * 65536 f16 fragment halfs, then N int32 indices
    _Float16* frag = (_Float16*)d_ws;
    int* idxi = (int*)((char*)d_ws + (size_t)25 * 65536 * sizeof(_Float16));

    // d_out: [pred E*N] [q N*E] [idx N] [sel N]
    float* out  = (float*)d_out;
    float* pred = out;
    float* qout = out + (size_t)E_EXP * N;
    float* idxf = qout + (size_t)N * E_EXP;
    float* sel  = idxf + N;

    repack_kernel<<<1024, 256, 0, stream>>>(Whid, M1, frag);

    decoder_kernel<<<dim3(N / ROWS_WG, E_EXP), NTHREADS, 0, stream>>>(
        P, Win, bin, bhid, frag, Wout, bout, pred, N);

    manager_kernel<<<N / ROWS_WG, NTHREADS, 0, stream>>>(
        P, M0, c0, frag + (size_t)24 * 65536, c1, M2, c2,
        qout, idxf, idxi, N);

    select_kernel<<<(N + 255) / 256, 256, 0, stream>>>(pred, idxi, sel, N);
}
